// CausalSelfAttention_78993038508480
// MI455X (gfx1250) — matrix operand order repaired and checked by reading, not compiled, not run on hardware
//
#include <hip/hip_runtime.h>
#include <hip/hip_bf16.h>

typedef __bf16 bf16;
typedef __attribute__((ext_vector_type(16))) __bf16 v16bf;
typedef __attribute__((ext_vector_type(8)))  __bf16 v8bf;
typedef __attribute__((ext_vector_type(8)))  float  v8f;

#define N_EMBD   2048
#define N_HEAD   16
#define HEAD_DIM 128
#define BATCH    4
#define SEQ      2048
#define C3       (3 * N_EMBD)
#define LPITCH   40     // GEMM LDS pitch (32 + 8 pad), 80B rows (16B aligned)
#define APITCH   136    // attention LDS pitch (128 + 8 pad), 272B rows (16B aligned)

// ---------------------------------------------------------------------------
// helpers
// ---------------------------------------------------------------------------
static __device__ __forceinline__ v16bf combine8(v8bf lo, v8bf hi) {
    v16bf r;
#pragma unroll
    for (int i = 0; i < 8; ++i) { r[i] = lo[i]; r[i + 8] = hi[i]; }
    return r;
}

// A-fragment (16x32 bf16) from row-major LDS tile [row][k].
// ISA layout: row = lane&15; K chunks [c0,c0+8) and [16+c0,16+c0+8), c0=(lane>>4)*8.
static __device__ __forceinline__ v16bf load_a_frag(const bf16* base, int pitch, int lane) {
    const int  row = lane & 15;
    const int  c0  = (lane >> 4) * 8;
    const bf16* p  = base + row * pitch;
    v8bf lo = *(const v8bf*)(p + c0);
    v8bf hi = *(const v8bf*)(p + 16 + c0);
    return combine8(lo, hi);
}

// B-fragment (32x16 bf16) from LDS tile stored K-contiguous per column [n][k].
// ISA layout: n = lane&15; K chunks [c0,c0+8) and [16+c0,16+c0+8), c0=(lane>>4)*8.
static __device__ __forceinline__ v16bf load_b_frag(const bf16* base, int pitch, int lane) {
    const int  n  = lane & 15;
    const int  k0 = (lane >> 4) * 8;
    const bf16* p = base + n * pitch + k0;
    v8bf lo = *(const v8bf*)(p);
    v8bf hi = *(const v8bf*)(p + 16);
    return combine8(lo, hi);
}

// CDNA5 async global->LDS copy, 16 bytes per lane, tracked by ASYNCcnt.
// For a generic pointer into __shared__, addr[31:0] is the LDS byte offset
// (aperture rules, CDNA5 ISA): feed it as the LDS-address
// VGPR of global_load_async_to_lds_b128.
static __device__ __forceinline__ void async_ld_b128(bf16* lds_dst, const bf16* g_src) {
    unsigned ldsoff = (unsigned)(size_t)lds_dst;
    asm volatile("global_load_async_to_lds_b128 %0, %1, off"
                 :
                 : "v"(ldsoff), "v"(g_src)
                 : "memory");
}
static __device__ __forceinline__ void wait_async0() {
    asm volatile("s_wait_asynccnt 0" ::: "memory");
}

// ---------------------------------------------------------------------------
// fp32 -> bf16 streaming convert
// ---------------------------------------------------------------------------
__global__ __launch_bounds__(256) void f32_to_bf16_kernel(const float* __restrict__ in,
                                                          bf16* __restrict__ out, int n) {
    int i = blockIdx.x * 1024 + threadIdx.x * 4;
    if (i + 3 < n) {
        float4 v = *(const float4*)(in + i);
        out[i + 0] = (bf16)v.x;
        out[i + 1] = (bf16)v.y;
        out[i + 2] = (bf16)v.z;
        out[i + 3] = (bf16)v.w;
    } else {
        for (int j = i; j < n; ++j) out[j] = (bf16)in[j];
    }
}

// ---------------------------------------------------------------------------
// Tiled WMMA GEMM: C[M,N] = A[M,K] @ B[K,N] + bias, bf16 in, f32 accum.
// Block tile 128x128, BK=32; 8 waves (4 M x 2 N), wave tile 32x64 = 2x4 WMMA.
// Register double-buffering: tile k+1 global loads overlap tile-k compute.
// ---------------------------------------------------------------------------
template <bool OUT_BF16>
__global__ __launch_bounds__(256) void gemm_bf16_kernel(const bf16* __restrict__ A,
                                                        const bf16* __restrict__ B,
                                                        const float* __restrict__ bias,
                                                        void* __restrict__ Cout,
                                                        int M, int N, int K) {
    __shared__ bf16 lA[128 * LPITCH];   // [m][k]
    __shared__ bf16 lB[128 * LPITCH];   // [n][k]  (transposed at stage time)

    const int tid  = threadIdx.x;
    const int lane = tid & 31;
    const int w    = tid >> 5;
    const int wm   = w & 3;    // 0..3  (M)
    const int wn   = w >> 2;   // 0..1  (N)
    const int n0   = blockIdx.x * 128;
    const int m0   = blockIdx.y * 128;

    // staging coordinates (2 chunks of 8 bf16 per thread for each of A and B)
    int arow[2], akc[2], bkk[2], bnc[2];
#pragma unroll
    for (int i = 0; i < 2; ++i) {
        int c   = tid + i * 256;
        arow[i] = c >> 2;
        akc[i]  = (c & 3) * 8;
        bkk[i]  = c >> 4;
        bnc[i]  = (c & 15) * 8;
    }

    v8f acc[2][4];
#pragma unroll
    for (int i = 0; i < 2; ++i)
#pragma unroll
        for (int j = 0; j < 4; ++j) acc[i][j] = v8f{};

    // prologue: fetch tile 0 into registers
    v8bf ra[2], rb[2];
#pragma unroll
    for (int i = 0; i < 2; ++i) {
        ra[i] = *(const v8bf*)(A + (size_t)(m0 + arow[i]) * K + akc[i]);
        rb[i] = *(const v8bf*)(B + (size_t)bkk[i] * N + n0 + bnc[i]);
    }

    for (int k0 = 0; k0 < K; k0 += 32) {
        // commit current tile registers to LDS
#pragma unroll
        for (int i = 0; i < 2; ++i) {
            *(v8bf*)(lA + arow[i] * LPITCH + akc[i]) = ra[i];
#pragma unroll
            for (int j = 0; j < 8; ++j) lB[(bnc[i] + j) * LPITCH + bkk[i]] = rb[i][j];
        }
        __syncthreads();

        // prefetch next tile from global (overlaps the WMMA chain below)
        if (k0 + 32 < K) {
#pragma unroll
            for (int i = 0; i < 2; ++i) {
                ra[i] = *(const v8bf*)(A + (size_t)(m0 + arow[i]) * K + (k0 + 32) + akc[i]);
                rb[i] = *(const v8bf*)(B + (size_t)(k0 + 32 + bkk[i]) * N + n0 + bnc[i]);
            }
        }

        // batch all fragment loads, then the 8-WMMA chain (amortized dscnt waits)
        v16bf afr[2], bfr[4];
        afr[0] = load_a_frag(lA + (wm * 32 + 0) * LPITCH, LPITCH, lane);
        afr[1] = load_a_frag(lA + (wm * 32 + 16) * LPITCH, LPITCH, lane);
#pragma unroll
        for (int ni = 0; ni < 4; ++ni)
            bfr[ni] = load_b_frag(lB + (wn * 64 + ni * 16) * LPITCH, LPITCH, lane);
#pragma unroll
        for (int ni = 0; ni < 4; ++ni) {
            acc[0][ni] = __builtin_amdgcn_wmma_f32_16x16x32_bf16(
                false, afr[0], false, bfr[ni], (short)0, acc[0][ni], false, false);
            acc[1][ni] = __builtin_amdgcn_wmma_f32_16x16x32_bf16(
                false, afr[1], false, bfr[ni], (short)0, acc[1][ni], false, false);
        }
        __syncthreads();
    }

    // epilogue: VGPR r holds row (r | r+8 by lane half), col = lane&15
    const int rbase = (lane >> 4) * 8;
    const int coll  = lane & 15;
#pragma unroll
    for (int mi = 0; mi < 2; ++mi)
#pragma unroll
        for (int ni = 0; ni < 4; ++ni) {
            int   col = n0 + wn * 64 + ni * 16 + coll;
            float bv  = bias[col];
#pragma unroll
            for (int r = 0; r < 8; ++r) {
                int   row = m0 + wm * 32 + mi * 16 + rbase + r;
                float v   = acc[mi][ni][r] + bv;
                if (OUT_BF16)
                    ((bf16*)Cout)[(size_t)row * N + col] = (bf16)v;
                else
                    ((float*)Cout)[(size_t)row * N + col] = v;
            }
        }
}

// ---------------------------------------------------------------------------
// Flash attention: one workgroup per (b, h, 128-row Q block). 8 waves, each
// owns 16 Q rows and a 16x128 fp32 accumulator.  KV blocks of 128 staged in
// LDS: K via CDNA5 async-to-LDS DMA (ASYNCcnt), V transposed via registers.
// ---------------------------------------------------------------------------
__global__ __launch_bounds__(256) void attn_kernel(const bf16* __restrict__ qkv,
                                                   bf16* __restrict__ attn_out) {
    __shared__ bf16 lK[128 * APITCH];        // [kvrow][d]   (async-filled)
    __shared__ bf16 lVt[128 * APITCH];       // [d][kvrow]
    __shared__ bf16 lP[8 * 16 * APITCH];     // per-wave P scratch [row][kvcol]

    const int tid  = threadIdx.x;
    const int lane = tid & 31;
    const int w    = tid >> 5;
    const int qb   = blockIdx.x & 15;        // T/128 = 16 q-blocks
    const int bh   = blockIdx.x >> 4;
    const int b    = bh >> 4;
    const int h    = bh & 15;
    const int q0   = qb * 128;
    const int qrow = q0 + w * 16;
    const float scale = 0.08838834764831845f;  // 1/sqrt(128)

    // Q fragments: 16 rows x 128 d as 4 A-frags, straight from global bf16
    v16bf qfr[4];
    {
        const int   row  = qrow + (lane & 15);
        const bf16* base = qkv + ((size_t)(b * SEQ + row)) * C3 + h * HEAD_DIM;
        const int   c0   = (lane >> 4) * 8;
#pragma unroll
        for (int f = 0; f < 4; ++f) {
            v8bf lo = *(const v8bf*)(base + f * 32 + c0);
            v8bf hi = *(const v8bf*)(base + f * 32 + 16 + c0);
            qfr[f]  = combine8(lo, hi);
        }
    }

    float mrow[8], lrow[8];
    v8f   O[8];
#pragma unroll
    for (int r = 0; r < 8; ++r) { mrow[r] = -1e30f; lrow[r] = 0.0f; }
#pragma unroll
    for (int nd = 0; nd < 8; ++nd) O[nd] = v8f{};

    bf16* myP = lP + w * 16 * APITCH;

    for (int kb = 0; kb <= qb; ++kb) {
        const int kv0 = kb * 128;
        // --- stage KV block: K via async DMA to LDS, V transposed via regs ---
#pragma unroll
        for (int i = 0; i < 8; ++i) {
            int    c    = tid + i * 256;
            int    r    = c >> 4;
            int    dc   = (c & 15) * 8;
            size_t grow = (size_t)(b * SEQ + kv0 + r) * C3 + h * HEAD_DIM + dc;
            async_ld_b128(lK + r * APITCH + dc, qkv + grow + N_EMBD);
            v8bf vv = *(const v8bf*)(qkv + grow + 2 * N_EMBD);
#pragma unroll
            for (int j = 0; j < 8; ++j) lVt[(dc + j) * APITCH + r] = vv[j];
        }
        wait_async0();
        __syncthreads();

        // --- scores: S(16x128) = Q @ K^T; batch 4 B-frags per 16-col tile ---
        v8f S[8];
#pragma unroll
        for (int n = 0; n < 8; ++n) {
            const bf16* krow = lK + (n * 16 + (lane & 15)) * APITCH + (lane >> 4) * 8;
            v16bf bfr[4];
#pragma unroll
            for (int f = 0; f < 4; ++f)
                bfr[f] = combine8(*(const v8bf*)(krow + f * 32),
                                  *(const v8bf*)(krow + f * 32 + 16));
            v8f s = v8f{};
#pragma unroll
            for (int f = 0; f < 4; ++f)
                s = __builtin_amdgcn_wmma_f32_16x16x32_bf16(
                    false, qfr[f], false, bfr[f], (short)0, s, false, false);
            if (kb == qb) {
                int col = kv0 + n * 16 + (lane & 15);
#pragma unroll
                for (int r = 0; r < 8; ++r) {
                    int rowa = qrow + r + ((lane >> 4) * 8);
                    s[r] = (col > rowa) ? -1e30f : s[r] * scale;
                }
            } else {
#pragma unroll
                for (int r = 0; r < 8; ++r) s[r] *= scale;
            }
            S[n] = s;
        }

        // --- online softmax; row M spans 16 lanes of a half-wave ---
#pragma unroll
        for (int r = 0; r < 8; ++r) {
            float mx = S[0][r];
#pragma unroll
            for (int n = 1; n < 8; ++n) mx = fmaxf(mx, S[n][r]);
#pragma unroll
            for (int off = 1; off < 16; off <<= 1) mx = fmaxf(mx, __shfl_xor(mx, off, 32));
            float mnew  = fmaxf(mrow[r], mx);
            float alpha = __expf(mrow[r] - mnew);
            mrow[r]     = mnew;
            float rs = 0.0f;
#pragma unroll
            for (int n = 0; n < 8; ++n) {
                float p = __expf(S[n][r] - mnew);
                S[n][r] = p;
                rs += p;
            }
#pragma unroll
            for (int off = 1; off < 16; off <<= 1) rs += __shfl_xor(rs, off, 32);
            lrow[r] = lrow[r] * alpha + rs;
#pragma unroll
            for (int nd = 0; nd < 8; ++nd) O[nd][r] *= alpha;
        }

        // --- P: C-layout -> per-wave LDS (bf16) -> A-layout fragments ---
        {
            const int colp  = lane & 15;
            const int rbase = (lane >> 4) * 8;
#pragma unroll
            for (int n = 0; n < 8; ++n)
#pragma unroll
                for (int r = 0; r < 8; ++r)
                    myP[(rbase + r) * APITCH + n * 16 + colp] = (bf16)S[n][r];
        }
        v16bf pfr[4];
        {
            const int   row  = lane & 15;
            const int   c0   = (lane >> 4) * 8;
            const bf16* base = myP + row * APITCH;
#pragma unroll
            for (int f = 0; f < 4; ++f)
                pfr[f] = combine8(*(const v8bf*)(base + f * 32 + c0),
                                  *(const v8bf*)(base + f * 32 + 16 + c0));
        }

        // --- O += P(16x128) @ V(128x128); batch 4 B-frags per d-tile ---
#pragma unroll
        for (int nd = 0; nd < 8; ++nd) {
            const bf16* vrow = lVt + (nd * 16 + (lane & 15)) * APITCH + (lane >> 4) * 8;
            v16bf bfr[4];
#pragma unroll
            for (int f = 0; f < 4; ++f)
                bfr[f] = combine8(*(const v8bf*)(vrow + f * 32),
                                  *(const v8bf*)(vrow + f * 32 + 16));
#pragma unroll
            for (int f = 0; f < 4; ++f)
                O[nd] = __builtin_amdgcn_wmma_f32_16x16x32_bf16(
                    false, pfr[f], false, bfr[f], (short)0, O[nd], false, false);
        }
        __syncthreads();   // protect lK/lVt before next stage
    }

    // --- epilogue: out = O / l, write bf16 [b*T+t][h*128+d] ---
    {
        const int colp  = lane & 15;
        const int rbase = (lane >> 4) * 8;
#pragma unroll
        for (int nd = 0; nd < 8; ++nd)
#pragma unroll
            for (int r = 0; r < 8; ++r) {
                float v   = O[nd][r] / lrow[r];
                int   row = qrow + rbase + r;
                attn_out[(size_t)(b * SEQ + row) * N_EMBD + h * HEAD_DIM + nd * 16 + colp] =
                    (bf16)v;
            }
    }
}

// ---------------------------------------------------------------------------
// launch
// ---------------------------------------------------------------------------
extern "C" void kernel_launch(void* const* d_in, const int* in_sizes, int n_in,
                              void* d_out, int out_size, void* d_ws, size_t ws_size,
                              hipStream_t stream) {
    const float* x     = (const float*)d_in[0];
    const float* Wqkv  = (const float*)d_in[1];
    const float* bqkv  = (const float*)d_in[2];
    const float* Wproj = (const float*)d_in[3];
    const float* bproj = (const float*)d_in[4];

    char* ws    = (char*)d_ws;
    bf16* xb    = (bf16*)ws;  ws += (size_t)BATCH * SEQ * N_EMBD * 2;
    bf16* wqkvb = (bf16*)ws;  ws += (size_t)N_EMBD * C3 * 2;
    bf16* wprojb= (bf16*)ws;  ws += (size_t)N_EMBD * N_EMBD * 2;
    bf16* qkvb  = (bf16*)ws;  ws += (size_t)BATCH * SEQ * C3 * 2;
    bf16* attnb = (bf16*)ws;

    const int rows = BATCH * SEQ;  // 8192

    int nx = rows * N_EMBD;
    f32_to_bf16_kernel<<<(nx + 1023) / 1024, 256, 0, stream>>>(x, xb, nx);
    int nw1 = N_EMBD * C3;
    f32_to_bf16_kernel<<<(nw1 + 1023) / 1024, 256, 0, stream>>>(Wqkv, wqkvb, nw1);
    int nw2 = N_EMBD * N_EMBD;
    f32_to_bf16_kernel<<<(nw2 + 1023) / 1024, 256, 0, stream>>>(Wproj, wprojb, nw2);

    // QKV projection: [8192,2048] @ [2048,6144] -> bf16 qkv (+bias)
    gemm_bf16_kernel<true><<<dim3(C3 / 128, rows / 128), 256, 0, stream>>>(
        xb, wqkvb, bqkv, qkvb, rows, C3, N_EMBD);

    // causal flash attention: 16 q-blocks x (B*H = 64) workgroups
    attn_kernel<<<dim3(16 * BATCH * N_HEAD), 256, 0, stream>>>(qkvb, attnb);

    // output projection: [8192,2048] @ [2048,2048] -> fp32 out (+bias)
    gemm_bf16_kernel<false><<<dim3(N_EMBD / 128, rows / 128), 256, 0, stream>>>(
        attnb, wprojb, bproj, d_out, rows, N_EMBD, N_EMBD);
}
